// DgaAllNet_50629074486131
// MI455X (gfx1250) — matrix operand order repaired and checked by reading, not compiled, not run on hardware
//
#include <hip/hip_runtime.h>
#include <cstddef>
#include <cstdint>

// ---------------------------------------------------------------------------
// CDNA5 WMMA plumbing (wave32, 16x16x32 f16 -> f32 accum)
// ---------------------------------------------------------------------------
typedef __attribute__((ext_vector_type(16))) _Float16 v16h;
typedef __attribute__((ext_vector_type(8)))  float    v8f;

union V16H { v16h v; float4 q[2]; _Float16 h[16]; };

__device__ __forceinline__ v8f wmma_f16(v16h a, v16h b, v8f c) {
  // 8 args: (neg_a, A, neg_b, B, c_mod, C, reuse_a, reuse_b)
  return __builtin_amdgcn_wmma_f32_16x16x32_f16(false, a, false, b, (short)0, c,
                                                false, false);
}

// A-fragment (16x32 f16) from row-major LDS [16][ld], K-tile base = kbase.
// Per ISA layout, a lane's 16 halfs are two contiguous 8-half runs:
//   K = kbase + 8*half + {0..7}  and  K = kbase + 16 + 8*half + {0..7}
// -> two ds_load_b128 per fragment.
__device__ __forceinline__ v16h lds_loadA(const _Float16* X, int ld, int kbase,
                                          int lane) {
  int m = lane & 15, half = lane >> 4;
  const _Float16* p = X + m * ld + kbase + half * 8;
  V16H u;
  u.q[0] = *(const float4*)(p);
  u.q[1] = *(const float4*)(p + 16);
  return u.v;
}

// B-fragment (32x16 f16) pre-packed so each lane reads 32 contiguous bytes.
__device__ __forceinline__ v16h load_packedB(const _Float16* p) {
  V16H u;
  u.q[0] = ((const float4*)p)[0];
  u.q[1] = ((const float4*)p)[1];
  return u.v;
}

__device__ __forceinline__ float sigf(float x) { return 1.0f / (1.0f + __expf(-x)); }
__device__ __forceinline__ float geluf(float x) {
  return 0.5f * x * (1.0f + erff(x * 0.70710678118654752f));
}

// ---------------------------------------------------------------------------
// Stage 0: edge-pad input  us[4][32][6] -> xpad[4][6][542]
// ---------------------------------------------------------------------------
__global__ void k_build_xpad(const float* __restrict__ us, float* __restrict__ xp) {
  const int total = 4 * 6 * 542;
  for (int idx = blockIdx.x * blockDim.x + threadIdx.x; idx < total;
       idx += gridDim.x * blockDim.x) {
    int p = idx % 542;
    int c = (idx / 542) % 6;
    int b = idx / (542 * 6);
    int s = p - 510; if (s < 0) s = 0;
    xp[idx] = us[(b * 32 + s) * 6 + c];
  }
}

// ---------------------------------------------------------------------------
// Stage 1: generic dilated conv1d + folded-BN + exact GELU
// ---------------------------------------------------------------------------
__global__ void k_conv1d(const float* __restrict__ x, int Cin, int Lin,
                         const float* __restrict__ w, const float* __restrict__ cb,
                         const float* __restrict__ bng, const float* __restrict__ bnb,
                         int dil, int ksz, int act,
                         float* __restrict__ y, int Cout) {
  int Lout = Lin - (ksz - 1) * dil;
  int total = 4 * Cout * Lout;
  for (int idx = blockIdx.x * blockDim.x + threadIdx.x; idx < total;
       idx += gridDim.x * blockDim.x) {
    int t  = idx % Lout;
    int co = (idx / Lout) % Cout;
    int b  = idx / (Lout * Cout);
    float s = cb[co];
    for (int ci = 0; ci < Cin; ++ci) {
      const float* xp = x + (b * Cin + ci) * Lin + t;
      const float* wp = w + (co * Cin + ci) * ksz;
      for (int j = 0; j < ksz; ++j) s += xp[j * dil] * wp[j];
    }
    if (act) {
      s = s * (bng[co] * rsqrtf(1.0f + 1e-5f)) + bnb[co];
      s = geluf(s);
    }
    y[idx] = s;
  }
}

// ---------------------------------------------------------------------------
// Stage 2: gyro calibration  w_hat = gyro_std*w_tilde + (I+C_w)@w_imu
//          y5[4][3][32], us[4][32][6] -> what[4][32][3]
// ---------------------------------------------------------------------------
__global__ void k_calib(const float* __restrict__ y5, const float* __restrict__ us,
                        const float* __restrict__ Cw, float* __restrict__ what) {
  const int total = 4 * 32 * 3;
  for (int idx = blockIdx.x * blockDim.x + threadIdx.x; idx < total;
       idx += gridDim.x * blockDim.x) {
    int i3 = idx % 3;
    int s  = (idx / 3) % 32;
    int b  = idx / 96;
    const float gstd = (i3 == 0) ? 0.017453292519943295f
                     : (i3 == 1) ? 0.034906585039886591f
                                 : 0.087266462599716474f;
    float wt = y5[(b * 3 + i3) * 32 + s];
    float acc = 0.0f;
    for (int j = 0; j < 3; ++j) {
      float cij = Cw[i3 * 3 + j] + ((i3 == j) ? 1.0f : 0.0f);
      acc += cij * us[(b * 32 + s) * 6 + j];
    }
    what[idx] = gstd * wt + acc;
  }
}

// ---------------------------------------------------------------------------
// Stage 3: precompute lw input projection over the 231 padded positions
//          xw[b][p][n] = lw_b[n] + sum_k what_pad[b][p][k] * lw_Wih[n][k]
// ---------------------------------------------------------------------------
__global__ void k_xw_pad(const float* __restrict__ what, const float* __restrict__ Wih,
                         const float* __restrict__ bias, float* __restrict__ xw) {
  const int total = 4 * 231 * 512;
  for (int idx = blockIdx.x * blockDim.x + threadIdx.x; idx < total;
       idx += gridDim.x * blockDim.x) {
    int n = idx & 511;
    int p = (idx >> 9) % 231;
    int b = idx / (231 * 512);
    int s = p - 199; if (s < 0) s = 0;
    const float* wh = what + (b * 32 + s) * 3;
    xw[idx] = bias[n] + wh[0] * Wih[n * 3 + 0] + wh[1] * Wih[n * 3 + 1]
                      + wh[2] * Wih[n * 3 + 2];
  }
}

// ---------------------------------------------------------------------------
// Stage 4: pack weights f32 -> f16 in exact WMMA B-fragment order.
// Fused K layout: kt < KxTiles comes from Wx[N][Kx] (zero-padded), the rest
// from Wh[N][Kh].  Packed element: out[(nt*KT+kt)*512 + lane*16 + e] with
// K = kt*32 + 8*(lane>>4) + (e&7) + 16*(e>>3),  N = nt*16 + (lane&15).
// ---------------------------------------------------------------------------
__global__ void k_pack_b(const float* __restrict__ Wx, int Kx, int KxTiles,
                         const float* __restrict__ Wh, int Kh,
                         int N, _Float16* __restrict__ out) {
  int KT = KxTiles + Kh / 32;
  int NT = N / 16;
  size_t total = (size_t)KT * NT * 512;
  for (size_t idx = (size_t)blockIdx.x * blockDim.x + threadIdx.x; idx < total;
       idx += (size_t)gridDim.x * blockDim.x) {
    int e    = (int)(idx & 15);
    int lane = (int)((idx >> 4) & 31);
    size_t tile = idx >> 9;
    int kt = (int)(tile % KT);
    int nt = (int)(tile / KT);
    int n = nt * 16 + (lane & 15);
    int k = kt * 32 + ((lane >> 4) << 3) + (e & 7) + ((e >> 3) << 4);
    float v;
    if (kt < KxTiles) {
      v = (k < Kx) ? Wx[(size_t)n * Kx + k] : 0.0f;
    } else {
      v = Wh[(size_t)n * Kh + (k - KxTiles * 32)];
    }
    out[idx] = (_Float16)v;
  }
}

// ---------------------------------------------------------------------------
// Stage 5: lw LSTM, all 32 windows in parallel (M = 128 sequences).
// 8 blocks x 4 waves; block owns an M-tile of 16 seqs; wave w owns hidden
// cols 32w..32w+31 across all four gates (8 N-tiles), K = 128 (4 K-tiles).
// acc is initialized from the precomputed xW table (includes bias).
// ---------------------------------------------------------------------------
__global__ __launch_bounds__(128, 1)
void k_lw(const _Float16* __restrict__ packW, const float* __restrict__ xwpad,
          _Float16* __restrict__ outw) {
  __shared__ _Float16 hbuf[16][128];
  int tid = threadIdx.x, lane = tid & 31, wv = tid >> 5, wg = blockIdx.x;
  int half = lane >> 4, nrow = lane & 15;

  for (int idx = tid; idx < 16 * 128; idx += 128)
    ((_Float16*)hbuf)[idx] = (_Float16)0.0f;
  float cst[2][8] = {};
  __syncthreads();

  for (int t = 0; t < 200; ++t) {
    v8f acc[4][2];
#pragma unroll
    for (int g = 0; g < 4; ++g)
#pragma unroll
      for (int tt = 0; tt < 2; ++tt) {
        int n = g * 128 + wv * 32 + tt * 16 + nrow;
#pragma unroll
        for (int r = 0; r < 8; ++r) {
          int m = r + 8 * half;
          int seq = wg * 16 + m;
          int win = seq >> 2, b = seq & 3;
          acc[g][tt][r] = xwpad[((size_t)b * 231 + (win + t)) * 512 + n];
        }
      }
    for (int kt = 0; kt < 4; ++kt) {
      v16h a = lds_loadA(&hbuf[0][0], 128, kt * 32, lane);
#pragma unroll
      for (int g = 0; g < 4; ++g)
#pragma unroll
        for (int tt = 0; tt < 2; ++tt) {
          int ntile = g * 8 + wv * 2 + tt;
          v16h bf = load_packedB(packW + ((size_t)(ntile * 4 + kt)) * 512 + lane * 16);
          acc[g][tt] = wmma_f16(a, bf, acc[g][tt]);
        }
    }
    __syncthreads();
#pragma unroll
    for (int tt = 0; tt < 2; ++tt)
#pragma unroll
      for (int r = 0; r < 8; ++r) {
        float c = sigf(acc[1][tt][r]) * cst[tt][r]
                + sigf(acc[0][tt][r]) * tanhf(acc[2][tt][r]);
        cst[tt][r] = c;
        float h = sigf(acc[3][tt][r]) * tanhf(c);
        int m = r + 8 * half;
        int hc = wv * 32 + tt * 16 + nrow;
        hbuf[m][hc] = (_Float16)h;
        int seq = wg * 16 + m;
        int win = seq >> 2, b = seq & 3;
        outw[(((size_t)win * 200 + t) * 4 + b) * 128 + hc] = (_Float16)h;
      }
    __syncthreads();
  }
}

// ---------------------------------------------------------------------------
// Stage 6: the serial scan.  One persistent 512-thread block (16 waves).
// LDS xh[16][768] f16 holds [out_w_t | out_q_t | h_lT].  lq (waves 0..3) and
// lT (all 16 waves) are interleaved per window step; FC + quaternion update
// close each of the 32 scan steps.
// ---------------------------------------------------------------------------
__global__ __launch_bounds__(512, 1)
void k_scan(const float* __restrict__ q0,
            const _Float16* __restrict__ packQ, const _Float16* __restrict__ packT,
            const float* __restrict__ lqWih, const float* __restrict__ lqb,
            const float* __restrict__ lTb, const _Float16* __restrict__ outw,
            const float* __restrict__ f0W, const float* __restrict__ f0b,
            const float* __restrict__ f1W, const float* __restrict__ f1b,
            const float* __restrict__ f2W, const float* __restrict__ f2b,
            float* __restrict__ out) {
  __shared__ _Float16 xh[16][768];
  __shared__ float qhist[33][16];   // [step][b*4+c]
  __shared__ float fc1[4][128];
  __shared__ float fc2[4][32];

  int tid = threadIdx.x, lane = tid & 31, wv = tid >> 5;
  int half = lane >> 4, nrow = lane & 15;

  if (tid < 16) {
    float v = q0[tid];
    qhist[0][tid] = v;
    out[(size_t)(tid >> 2) * 33 * 4 + (tid & 3)] = v;   // qs[:,0] = q0
  }
  __syncthreads();

  for (int i = 0; i < 32; ++i) {
    // fresh LSTM states: zero lq-h (cols 128..255) and lT-h (cols 256..767)
    for (int idx = tid; idx < 16 * 640; idx += 512)
      xh[idx / 640][128 + idx % 640] = (_Float16)0.0f;
    float cq[2][8] = {};
    float cT[2][8] = {};
    __syncthreads();

    for (int t = 0; t < 200; ++t) {
      // (A) stream out_w_t into xh cols 0..127 (rows >=4 are padding)
      if (tid < 256) {
        int row = tid >> 4;
        int col = (tid & 15) * 8;
        float4 v;
        if (row < 4)
          v = *(const float4*)(outw + (((size_t)i * 200 + t) * 4 + row) * 128 + col);
        else
          v = make_float4(0.f, 0.f, 0.f, 0.f);
        *(float4*)&xh[row][col] = v;
      }

      // (B1) lq GEMM on waves 0..3; x-part (4-dim) folded into acc init
      v8f accq[4][2] = {};
      if (wv < 4) {
        int j = t - (199 - i); if (j < 0) j = 0;   // q_buf row -> q_j
#pragma unroll
        for (int g = 0; g < 4; ++g)
#pragma unroll
          for (int tt = 0; tt < 2; ++tt) {
            int n = g * 128 + wv * 32 + tt * 16 + nrow;
            float base = lqb[n];
            float w0 = lqWih[n * 4 + 0], w1 = lqWih[n * 4 + 1];
            float w2 = lqWih[n * 4 + 2], w3 = lqWih[n * 4 + 3];
#pragma unroll
            for (int r = 0; r < 8; ++r) {
              int m = r + 8 * half;
              float s = base;
              if (m < 4) {
                const float* qv = &qhist[j][m * 4];
                s += qv[0] * w0 + qv[1] * w1 + qv[2] * w2 + qv[3] * w3;
              }
              accq[g][tt][r] = s;
            }
          }
        for (int kt = 0; kt < 4; ++kt) {
          v16h a = lds_loadA(&xh[0][0], 768, 128 + kt * 32, lane);
#pragma unroll
          for (int g = 0; g < 4; ++g)
#pragma unroll
            for (int tt = 0; tt < 2; ++tt) {
              int ntile = g * 8 + wv * 2 + tt;
              v16h bf = load_packedB(packQ + ((size_t)(ntile * 4 + kt)) * 512 + lane * 16);
              accq[g][tt] = wmma_f16(a, bf, accq[g][tt]);
            }
        }
      }
      __syncthreads();

      // (B2) lq activations; h_lq == out_q_t, written straight into xh 128..255
      if (wv < 4) {
#pragma unroll
        for (int tt = 0; tt < 2; ++tt)
#pragma unroll
          for (int r = 0; r < 8; ++r) {
            float c = sigf(accq[1][tt][r]) * cq[tt][r]
                    + sigf(accq[0][tt][r]) * tanhf(accq[2][tt][r]);
            cq[tt][r] = c;
            float h = sigf(accq[3][tt][r]) * tanhf(c);
            int m = r + 8 * half;
            int hc = wv * 32 + tt * 16 + nrow;
            xh[m][128 + hc] = (_Float16)h;
          }
      }
      __syncthreads();

      // (C) lT fused GEMM: [16 x 768] x [768 x 2048], all 16 waves
      v8f acc[4][2];
#pragma unroll
      for (int g = 0; g < 4; ++g)
#pragma unroll
        for (int tt = 0; tt < 2; ++tt) {
          int n = g * 512 + wv * 32 + tt * 16 + nrow;
          float bb = lTb[n];
#pragma unroll
          for (int r = 0; r < 8; ++r) acc[g][tt][r] = bb;
        }
      for (int kt = 0; kt < 24; ++kt) {
        v16h a = lds_loadA(&xh[0][0], 768, kt * 32, lane);
#pragma unroll
        for (int g = 0; g < 4; ++g)
#pragma unroll
          for (int tt = 0; tt < 2; ++tt) {
            int ntile = g * 32 + wv * 2 + tt;
            v16h bf = load_packedB(packT + ((size_t)(ntile * 24 + kt)) * 512 + lane * 16);
            acc[g][tt] = wmma_f16(a, bf, acc[g][tt]);
          }
      }
      __syncthreads();

      // (D) lT activations; broadcast new h into xh cols 256..767
#pragma unroll
      for (int tt = 0; tt < 2; ++tt)
#pragma unroll
        for (int r = 0; r < 8; ++r) {
          float c = sigf(acc[1][tt][r]) * cT[tt][r]
                  + sigf(acc[0][tt][r]) * tanhf(acc[2][tt][r]);
          cT[tt][r] = c;
          float h = sigf(acc[3][tt][r]) * tanhf(c);
          int m = r + 8 * half;
          int hc = wv * 32 + tt * 16 + nrow;
          xh[m][256 + hc] = (_Float16)h;
        }
      // next iteration's first barrier orders these writes before the GEMM
    }
    __syncthreads();

    // FC chain (pure linear in the reference): 512 -> 128 -> 32 -> 4
    {
      int b = tid >> 7, o = tid & 127;   // exactly 4*128 = 512 threads
      float s = f0b[o];
      const float* wr = f0W + (size_t)o * 512;
      for (int k = 0; k < 512; ++k) s += (float)xh[b][256 + k] * wr[k];
      fc1[b][o] = s;
    }
    __syncthreads();
    if (tid < 128) {
      int b = tid >> 5, o = tid & 31;
      float s = f1b[o];
      const float* wr = f1W + o * 128;
      for (int k = 0; k < 128; ++k) s += fc1[b][k] * wr[k];
      fc2[b][o] = s;
    }
    __syncthreads();
    if (tid < 4) {
      int b = tid;
      float qn[4]; float ssum = 0.0f;
      for (int c = 0; c < 4; ++c) {
        float s = f2b[c];
        const float* wr = f2W + c * 32;
        for (int k = 0; k < 32; ++k) s += fc2[b][k] * wr[k];
        float v = qhist[i][b * 4 + c] + s;
        qn[c] = v; ssum += v;
      }
      for (int c = 0; c < 4; ++c) {
        float v = qn[c] / ssum;
        qhist[i + 1][b * 4 + c] = v;
        out[((size_t)b * 33 + i + 1) * 4 + c] = v;
      }
    }
    __syncthreads();
  }
}

// ---------------------------------------------------------------------------
// Host-side orchestration
// ---------------------------------------------------------------------------
extern "C" void kernel_launch(void* const* d_in, const int* in_sizes, int n_in,
                              void* d_out, int out_size, void* d_ws, size_t ws_size,
                              hipStream_t stream) {
  (void)in_sizes; (void)n_in; (void)out_size; (void)ws_size;

  // setup_inputs order: us, q0, then params dict in insertion order
  const float* us      = (const float*)d_in[0];
  const float* q0      = (const float*)d_in[1];
  const float* conv1_w = (const float*)d_in[2];
  const float* conv1_b = (const float*)d_in[3];
  const float* conv2_w = (const float*)d_in[4];
  const float* conv2_b = (const float*)d_in[5];
  const float* conv3_w = (const float*)d_in[6];
  const float* conv3_b = (const float*)d_in[7];
  const float* conv4_w = (const float*)d_in[8];
  const float* conv4_b = (const float*)d_in[9];
  const float* conv5_w = (const float*)d_in[10];
  const float* conv5_b = (const float*)d_in[11];
  const float* bn1_g   = (const float*)d_in[12];
  const float* bn1_b   = (const float*)d_in[13];
  const float* bn2_g   = (const float*)d_in[14];
  const float* bn2_b   = (const float*)d_in[15];
  const float* bn3_g   = (const float*)d_in[16];
  const float* bn3_b   = (const float*)d_in[17];
  const float* bn4_g   = (const float*)d_in[18];
  const float* bn4_b   = (const float*)d_in[19];
  const float* C_w     = (const float*)d_in[20];
  const float* lw_Wih  = (const float*)d_in[21];
  const float* lw_Whh  = (const float*)d_in[22];
  const float* lw_b    = (const float*)d_in[23];
  const float* lq_Wih  = (const float*)d_in[24];
  const float* lq_Whh  = (const float*)d_in[25];
  const float* lq_b    = (const float*)d_in[26];
  const float* lT_Wih  = (const float*)d_in[27];
  const float* lT_Whh  = (const float*)d_in[28];
  const float* lT_b    = (const float*)d_in[29];
  const float* f0_W    = (const float*)d_in[30];
  const float* f0_b    = (const float*)d_in[31];
  const float* f1_W    = (const float*)d_in[32];
  const float* f1_b    = (const float*)d_in[33];
  const float* f2_W    = (const float*)d_in[34];
  const float* f2_b    = (const float*)d_in[35];

  char* ws = (char*)d_ws;
  size_t off = 0;
  auto carve = [&](size_t bytes) -> void* {
    void* p = ws + off;
    off += (bytes + 255) & ~(size_t)255;
    return p;
  };
  float*    xpad   = (float*)carve((size_t)4 * 6 * 542 * 4);
  float*    y1     = (float*)carve((size_t)4 * 16 * 536 * 4);
  float*    y2     = (float*)carve((size_t)4 * 32 * 512 * 4);
  float*    y3     = (float*)carve((size_t)4 * 64 * 416 * 4);
  float*    y4     = (float*)carve((size_t)4 * 128 * 32 * 4);
  float*    y5     = (float*)carve((size_t)4 * 3 * 32 * 4);
  float*    what   = (float*)carve((size_t)4 * 32 * 3 * 4);
  float*    xwpad  = (float*)carve((size_t)4 * 231 * 512 * 4);
  _Float16* packLW = (_Float16*)carve((size_t)32 * 4 * 512 * 2);
  _Float16* packLQ = (_Float16*)carve((size_t)32 * 4 * 512 * 2);
  _Float16* packLT = (_Float16*)carve((size_t)128 * 24 * 512 * 2);
  _Float16* outw   = (_Float16*)carve((size_t)32 * 200 * 4 * 128 * 2);

  // GyroNet conv stack (parallel, trivial FLOPs)
  k_build_xpad<<<32, 256, 0, stream>>>(us, xpad);
  k_conv1d<<<128, 256, 0, stream>>>(xpad, 6, 542, conv1_w, conv1_b, bn1_g, bn1_b, 1, 7, 1, y1, 16);
  k_conv1d<<<128, 256, 0, stream>>>(y1, 16, 536, conv2_w, conv2_b, bn2_g, bn2_b, 4, 7, 1, y2, 32);
  k_conv1d<<<192, 256, 0, stream>>>(y2, 32, 512, conv3_w, conv3_b, bn3_g, bn3_b, 16, 7, 1, y3, 64);
  k_conv1d<<<64, 256, 0, stream>>>(y3, 64, 416, conv4_w, conv4_b, bn4_g, bn4_b, 64, 7, 1, y4, 128);
  k_conv1d<<<8, 256, 0, stream>>>(y4, 128, 32, conv5_w, conv5_b, nullptr, nullptr, 1, 1, 0, y5, 3);
  k_calib<<<2, 256, 0, stream>>>(y5, us, C_w, what);
  k_xw_pad<<<256, 256, 0, stream>>>(what, lw_Wih, lw_b, xwpad);

  // One-time weight repack into WMMA B-fragment order (f16)
  k_pack_b<<<64, 256, 0, stream>>>(nullptr, 0, 0, lw_Whh, 128, 512, packLW);
  k_pack_b<<<64, 256, 0, stream>>>(nullptr, 0, 0, lq_Whh, 128, 512, packLQ);
  k_pack_b<<<512, 256, 0, stream>>>(lT_Wih, 256, 8, lT_Whh, 512, 2048, packLT);

  // lw LSTM: 32 independent windows batched (M = 128)
  k_lw<<<8, 128, 0, stream>>>(packLW, xwpad, outw);

  // Serial scan: persistent single workgroup, lq+lT interleaved, FC, q update
  k_scan<<<1, 512, 0, stream>>>(q0, packLQ, packLT, lq_Wih, lq_b, lT_b, outw,
                                f0_W, f0_b, f1_W, f1_b, f2_W, f2_b,
                                (float*)d_out);
}
